// WITFormerBlock_29978871726221
// MI455X (gfx1250) — matrix operand order repaired and checked by reading, not compiled, not run on hardware
//
#include <hip/hip_runtime.h>
#include <hip/hip_bf16.h>

typedef _Float16 half8 __attribute__((ext_vector_type(8)));
typedef _Float16 v16h  __attribute__((ext_vector_type(16)));
typedef float    v8f   __attribute__((ext_vector_type(8)));
typedef float    vf4   __attribute__((ext_vector_type(4)));

#define NH_   8
#define HD_   32
#define C_    256
#define NTOK  64
#define SX    264          // padded half-stride for x / attn-out tile
#define LP    72           // padded half-stride for P row-block staging

// LDS layout (bytes) -- total 157728 <= 160KB => 2 workgroups / WGP
#define OFF_X    0                       // 64*264*2        = 33792
#define OFF_Q    33792                   // 8*64*32*2       = 32768
#define OFF_K    66560                   // 32768
#define OFF_VT   99328                   // 32768 (V transposed [h][d][tok])
#define OFF_P    132096                  // 8*16*72*2       = 18432 (row-block only)
#define OFF_RPB  150528                  // 225*8*4         = 7200
#define SMEM_BYTES 157728

__device__ __forceinline__ v8f wmma_f16(v16h a, v16h b, v8f c) {
  return __builtin_amdgcn_wmma_f32_16x16x32_f16(false, a, false, b, (short)0, c,
                                                false, false);
}

// A fragment (16x32 f16, M x K): lane holds row M=lane%16;
// halves e=0..7 -> K = k0 + 8*(lane/16) + e ; e=8..15 -> K += 16.
__device__ __forceinline__ v16h frag_a(const _Float16* tile, int ld, int lane) {
  union { v16h v; half8 h[2]; } u;
  const _Float16* p = tile + (lane & 15) * ld + ((lane >> 4) << 3);
  u.h[0] = *(const half8*)(p);
  u.h[1] = *(const half8*)(p + 16);
  return u.v;
}

// B fragment (32x16 f16, K x N) from B^T storage bt[n][k]:
// lane holds column N=lane%16; e=0..7 -> K = k0 + 8*(lane/16) + e ; e=8..15 -> K += 16.
__device__ __forceinline__ v16h frag_b(const _Float16* bt, int ld, int lane) {
  union { v16h v; half8 h[2]; } u;
  const _Float16* p = bt + (lane & 15) * ld + ((lane >> 4) << 3);
  u.h[0] = *(const half8*)(p);
  u.h[1] = *(const half8*)(p + 16);
  return u.v;
}

// 64x32 = [64 x KDIM] * [KDIM x 32] accumulate; BT is [32 rows][KDIM]
template <int KDIM>
__device__ __forceinline__ void gemm_64x32(const _Float16* A, int lda,
                                           const _Float16* BT, int ldb,
                                           int lane, v8f acc[4][2]) {
#pragma unroll
  for (int k0 = 0; k0 < KDIM; k0 += 32) {
    v16h b0 = frag_b(BT + k0, ldb, lane);
    v16h b1 = frag_b(BT + 16 * ldb + k0, ldb, lane);
#pragma unroll
    for (int mi = 0; mi < 4; ++mi) {
      v16h a = frag_a(A + (mi * 16) * lda + k0, lda, lane);
      acc[mi][0] = wmma_f16(a, b0, acc[mi][0]);
      acc[mi][1] = wmma_f16(a, b1, acc[mi][1]);
    }
  }
}

// ---- prep: fp32 [K][N] row-major -> f16 transposed [N][K] ----
__global__ void transpose_to_f16_kernel(const float* __restrict__ src,
                                        _Float16* __restrict__ dst,
                                        int K, int N) {
  int i = blockIdx.x * blockDim.x + threadIdx.x;
  if (i >= K * N) return;
  int k = i % K;
  int n = i / K;
  dst[n * K + k] = (_Float16)src[k * N + n];
}

// ---- fused window attention: one block per 8x8 window, wave w == head w ----
__global__ __launch_bounds__(256)
void swin_window_attn_kernel(const float* __restrict__ x,
                             const float* __restrict__ bq,
                             const float* __restrict__ bkv,
                             const float* __restrict__ rpb_table,
                             const float* __restrict__ proj_b,
                             const _Float16* __restrict__ wqT,
                             const _Float16* __restrict__ wkvT,
                             const _Float16* __restrict__ projT,
                             float* __restrict__ out) {
  extern __shared__ char smem[];
  _Float16* s_x   = (_Float16*)(smem + OFF_X);    // [64][SX] x tile, later attn-out
  _Float16* s_q   = (_Float16*)(smem + OFF_Q);    // [8][64][32]
  _Float16* s_k   = (_Float16*)(smem + OFF_K);    // [8][64][32]
  _Float16* s_vt  = (_Float16*)(smem + OFF_VT);   // [8][32][64] (transposed V)
  _Float16* s_p   = (_Float16*)(smem + OFF_P);    // [8][16][LP] row-block staging
  float*    s_rpb = (float*)   (smem + OFF_RPB);  // [225][8]

  const int tid  = threadIdx.x;
  const int lane = tid & 31;
  const int h    = tid >> 5;            // wave id == head id (NH_ == 8 waves)
  const int rowh = (lane >> 4) << 3;    // C/D row offset: 0 or 8
  const int col  = lane & 15;           // C/D column

  const int bidx = blockIdx.x;          // 4096 windows
  const int b  = bidx >> 10;
  const int win = bidx & 1023;
  const int wy = win >> 5;
  const int wx = win & 31;

  const float SCALE = 0.17677669529663687f; // 1/sqrt(32)

  // keep weights hot in L2 (global_prefetch)
  __builtin_prefetch(wqT  + (size_t)h * HD_ * C_, 0, 3);
  __builtin_prefetch(wkvT + (size_t)h * HD_ * C_, 0, 3);
  __builtin_prefetch(wkvT + (size_t)(C_ + h * HD_) * C_, 0, 3);
  __builtin_prefetch(projT + (size_t)h * HD_ * C_, 0, 3);

  // ---- stage 0/1: load rpb table + x window -> f16 LDS [tok][cin] ----
  for (int i = tid; i < 225 * NH_; i += 256) s_rpb[i] = rpb_table[i];

  // x is touched exactly once per element: stream it non-temporally so L2
  // stays dedicated to the weight slices all 4096 blocks share.
  const float* xb = x + (size_t)b * C_ * 65536 + (size_t)(wy * 8) * 256 + wx * 8;
#pragma unroll
  for (int it = 0; it < 8; ++it) {
    int p = tid + it * 256;             // 2048 (channel,row) pairs
    int c = p >> 3;
    int r = p & 7;
    const float* px = xb + (size_t)c * 65536 + r * 256;
    vf4 f0 = __builtin_nontemporal_load((const vf4*)px);
    vf4 f1 = __builtin_nontemporal_load(((const vf4*)px) + 1);
    _Float16* dst = s_x + (r * 8) * SX + c;   // tokens r*8+0..7, channel c
    dst[0 * SX] = (_Float16)f0[0];
    dst[1 * SX] = (_Float16)f0[1];
    dst[2 * SX] = (_Float16)f0[2];
    dst[3 * SX] = (_Float16)f0[3];
    dst[4 * SX] = (_Float16)f1[0];
    dst[5 * SX] = (_Float16)f1[1];
    dst[6 * SX] = (_Float16)f1[2];
    dst[7 * SX] = (_Float16)f1[3];
  }
  __syncthreads();

  _Float16* qh  = s_q  + (size_t)h * NTOK * HD_;
  _Float16* kh  = s_k  + (size_t)h * NTOK * HD_;
  _Float16* vth = s_vt + (size_t)h * HD_ * NTOK;
  _Float16* ph  = s_p  + (size_t)h * 16 * LP;   // one 16-row P block per head

  // ---- stage 2: per-head Q, K, V projections via WMMA ----
  { // Q = (xw @ wq + bq) * scale  -> s_q[h][tok][d]
    v8f acc[4][2] = {};
    gemm_64x32<C_>(s_x, SX, wqT + (size_t)(h * HD_) * C_, C_, lane, acc);
#pragma unroll
    for (int ni = 0; ni < 2; ++ni) {
      float bv = bq[h * HD_ + ni * 16 + col];
#pragma unroll
      for (int mi = 0; mi < 4; ++mi)
#pragma unroll
        for (int r = 0; r < 8; ++r) {
          int tok = mi * 16 + rowh + r;
          qh[tok * HD_ + ni * 16 + col] = (_Float16)((acc[mi][ni][r] + bv) * SCALE);
        }
    }
  }
  { // K -> s_k[h][tok][d]
    v8f acc[4][2] = {};
    gemm_64x32<C_>(s_x, SX, wkvT + (size_t)(h * HD_) * C_, C_, lane, acc);
#pragma unroll
    for (int ni = 0; ni < 2; ++ni) {
      float bv = bkv[h * HD_ + ni * 16 + col];
#pragma unroll
      for (int mi = 0; mi < 4; ++mi)
#pragma unroll
        for (int r = 0; r < 8; ++r) {
          int tok = mi * 16 + rowh + r;
          kh[tok * HD_ + ni * 16 + col] = (_Float16)(acc[mi][ni][r] + bv);
        }
    }
  }
  { // V -> transposed s_vt[h][d][tok] (ready as GEMM2 B^T operand)
    v8f acc[4][2] = {};
    gemm_64x32<C_>(s_x, SX, wkvT + (size_t)(C_ + h * HD_) * C_, C_, lane, acc);
#pragma unroll
    for (int ni = 0; ni < 2; ++ni) {
      float bv = bkv[C_ + h * HD_ + ni * 16 + col];
#pragma unroll
      for (int mi = 0; mi < 4; ++mi)
#pragma unroll
        for (int r = 0; r < 8; ++r) {
          int tok = mi * 16 + rowh + r;
          vth[(ni * 16 + col) * NTOK + tok] = (_Float16)(acc[mi][ni][r] + bv);
        }
    }
  }
  __syncthreads();   // everyone done reading s_x before it is reused as attn-out

  // ---- stage 3: attention, streamed one 16-row block at a time ----
  // S row-block = Q(mi) K^T ; softmax ; P(mi) staged in LDS ; O(mi) = P(mi) V
#pragma unroll 1
  for (int mi = 0; mi < 4; ++mi) {
    v16h qf = frag_a(qh + mi * 16 * HD_, HD_, lane);   // K = 32 in one WMMA
    v8f s[4];
#pragma unroll
    for (int ni = 0; ni < 4; ++ni) {
      v16h kf = frag_b(kh + ni * 16 * HD_, HD_, lane); // B = K^T (rows = key tok)
      v8f z = {};
      s[ni] = wmma_f16(qf, kf, z);
    }
    // relative position bias, computed inline from LDS table
#pragma unroll
    for (int ni = 0; ni < 4; ++ni) {
      int j = ni * 16 + col, yj = j >> 3, xj = j & 7;
#pragma unroll
      for (int r = 0; r < 8; ++r) {
        int i = mi * 16 + rowh + r, yi = i >> 3, xi = i & 7;
        int idx = (yi - yj + 7) * 15 + (xi - xj + 7);
        s[ni][r] += s_rpb[idx * NH_ + h];
      }
    }
    // row softmax over 64 cols: 4 tiles in-lane + shuffle over 16 lanes/half
    float mx[8], sum[8];
#pragma unroll
    for (int r = 0; r < 8; ++r)
      mx[r] = fmaxf(fmaxf(s[0][r], s[1][r]), fmaxf(s[2][r], s[3][r]));
#pragma unroll
    for (int off = 1; off < 16; off <<= 1)
#pragma unroll
      for (int r = 0; r < 8; ++r)
        mx[r] = fmaxf(mx[r], __shfl_xor(mx[r], off, 16));
#pragma unroll
    for (int r = 0; r < 8; ++r) sum[r] = 0.0f;
#pragma unroll
    for (int ni = 0; ni < 4; ++ni)
#pragma unroll
      for (int r = 0; r < 8; ++r) {
        float e = __expf(s[ni][r] - mx[r]);
        s[ni][r] = e;
        sum[r] += e;
      }
#pragma unroll
    for (int off = 1; off < 16; off <<= 1)
#pragma unroll
      for (int r = 0; r < 8; ++r)
        sum[r] += __shfl_xor(sum[r], off, 16);
#pragma unroll
    for (int r = 0; r < 8; ++r) sum[r] = 1.0f / sum[r];
    // stage 16-row P block (same-wave LDS ops are in order)
#pragma unroll
    for (int ni = 0; ni < 4; ++ni)
#pragma unroll
      for (int r = 0; r < 8; ++r)
        ph[(rowh + r) * LP + ni * 16 + col] = (_Float16)(s[ni][r] * sum[r]);

    // O(mi) = P(mi)[16x64] @ V[64x32], two chained K-steps per output tile
    v16h a0 = frag_a(ph, LP, lane);        // keys 0..31
    v16h a1 = frag_a(ph + 32, LP, lane);   // keys 32..63
#pragma unroll
    for (int ni = 0; ni < 2; ++ni) {
      v8f z = {};
      v8f t = wmma_f16(a0, frag_b(vth + ni * 16 * NTOK, NTOK, lane), z);
      t = wmma_f16(a1, frag_b(vth + ni * 16 * NTOK + 32, NTOK, lane), t);
#pragma unroll
      for (int r = 0; r < 8; ++r)
        s_x[(mi * 16 + rowh + r) * SX + h * HD_ + ni * 16 + col] = (_Float16)t[r];
    }
  }
  __syncthreads();   // attn-out complete across all heads

  // ---- stage 4: output projection + window reverse, wave w owns cout block ----
  {
    v8f acc[4][2] = {};
    gemm_64x32<C_>(s_x, SX, projT + (size_t)(h * HD_) * C_, C_, lane, acc);
    const int hbase = wy * 8, wbase = wx * 8;
#pragma unroll
    for (int ni = 0; ni < 2; ++ni) {
      int n = h * HD_ + ni * 16 + col;
      float bv = proj_b[n];
      float* op = out + ((size_t)(b * C_ + n) * 256 + hbase) * 256 + wbase;
#pragma unroll
      for (int mi = 0; mi < 4; ++mi)
#pragma unroll
        for (int r = 0; r < 8; ++r) {
          int tok = mi * 16 + rowh + r;
          // out written exactly once: stream past L2 with NT stores
          __builtin_nontemporal_store(acc[mi][ni][r] + bv,
                                      op + (tok >> 3) * 256 + (tok & 7));
        }
    }
  }
}

extern "C" void kernel_launch(void* const* d_in, const int* in_sizes, int n_in,
                              void* d_out, int out_size, void* d_ws, size_t ws_size,
                              hipStream_t stream) {
  const float* x   = (const float*)d_in[0];
  const float* wq  = (const float*)d_in[1];
  const float* bq  = (const float*)d_in[2];
  const float* wkv = (const float*)d_in[3];
  const float* bkv = (const float*)d_in[4];
  const float* rpb = (const float*)d_in[5];
  const float* pw  = (const float*)d_in[6];
  const float* pb  = (const float*)d_in[7];
  float* out = (float*)d_out;

  // workspace: f16 transposed weights
  _Float16* wqT   = (_Float16*)d_ws;                           // 256*256 halves
  _Float16* wkvT  = (_Float16*)((char*)d_ws + 131072);         // 512*256 halves
  _Float16* projT = (_Float16*)((char*)d_ws + 393216);         // 256*256 halves

  transpose_to_f16_kernel<<<(256 * 256 + 255) / 256, 256, 0, stream>>>(wq, wqT, 256, 256);
  transpose_to_f16_kernel<<<(256 * 512 + 255) / 256, 256, 0, stream>>>(wkv, wkvT, 256, 512);
  transpose_to_f16_kernel<<<(256 * 256 + 255) / 256, 256, 0, stream>>>(pw, projT, 256, 256);

  // 4 batches * 32*32 windows = 4096 blocks, 256 threads (8 waves = 8 heads)
  swin_window_attn_kernel<<<4096, 256, SMEM_BYTES, stream>>>(
      x, bq, bkv, rpb, pb, wqT, wkvT, projT, out);
}
